// NormalizedMultiHeadAttention_2190433321585
// MI455X (gfx1250) — compile-verified, matrix operand order checked by reading, not run on hardware
//
#include <hip/hip_runtime.h>
#include <math.h>

// ---------------------------------------------------------------------------
// NormalizedMultiHeadAttention forward for gfx1250 (MI455X)
// f16 WMMA pipeline + Tensor Data Mover (TDM) LDS staging in the GEMMs
// ---------------------------------------------------------------------------

typedef __attribute__((ext_vector_type(16))) _Float16 v16h;
typedef __attribute__((ext_vector_type(8)))  _Float16 v8h;
typedef __attribute__((ext_vector_type(8)))  float    v8f;
typedef __attribute__((ext_vector_type(4)))  unsigned u32x4;
typedef __attribute__((ext_vector_type(8)))  unsigned u32x8;

#define BATCH   2
#define SEQ     2048
#define DMODEL  2048
#define HEADS   16
#define HDIM    128

// ---------------------------------------------------------------------------
// WMMA helper: D = A(16x32 f16) * B(32x16 f16) + C(16x16 f32)
// ---------------------------------------------------------------------------
__device__ __forceinline__ v8f wmma_f16(v16h a, v16h b, v8f c) {
    return __builtin_amdgcn_wmma_f32_16x16x32_f16(
        false, a, false, b, (short)0, c, false, false);
}

// ---------------------------------------------------------------------------
// Fragment loader from global, row-major [R][ld] (K contiguous).
//   lane l<16 : row row0+l,    halves 0-7 -> K k0..k0+7,  8-15 -> k0+16..+23
//   lane l>=16: row row0+l-16, halves 0-7 -> K k0+8..+15, 8-15 -> k0+24..+31
// Serves both the A-fragment (rows) and B-fragment (columns of [N][K] store).
// ---------------------------------------------------------------------------
__device__ __forceinline__ v16h load_frag(const _Float16* __restrict__ p,
                                          int ld, int row0, int k0, int lane) {
    int r  = row0 + (lane & 15);
    int kb = k0 + ((lane >> 4) << 3);
    const _Float16* q = p + (size_t)r * ld + kb;
    v8h lo = *(const v8h*)(q);
    v8h hi = *(const v8h*)(q + 16);
    v16h f;
#pragma unroll
    for (int i = 0; i < 8; ++i) { f[i] = lo[i]; f[i + 8] = hi[i]; }
    return f;
}

// Same fragment math against an LDS tile stored [rows][32] f16.
__device__ __forceinline__ v16h load_frag_lds(const _Float16* p,
                                              int row0, int lane) {
    int r  = row0 + (lane & 15);
    int kb = (lane >> 4) << 3;
    const _Float16* q = p + r * 32 + kb;
    v8h lo = *(const v8h*)(q);
    v8h hi = *(const v8h*)(q + 16);
    v16h f;
#pragma unroll
    for (int i = 0; i < 8; ++i) { f[i] = lo[i]; f[i + 8] = hi[i]; }
    return f;
}

// ---------------------------------------------------------------------------
// TDM descriptor helpers (CDNA5 ISA).
// 2-D tile: tile_dim0 = 32 elems along K (contiguous), tile_dim1 = rows,
// tensor_dim0_stride = 2048 elems, data_size = 2 bytes, LDS packed [rows][32].
// ---------------------------------------------------------------------------
__device__ __forceinline__ u32x4 tdm_group0(unsigned lds_off,
                                            const _Float16* gaddr) {
    unsigned long long ga = (unsigned long long)(uintptr_t)gaddr;
    u32x4 g;
    g.x = 1u;                                      // count=1, user descriptor
    g.y = lds_off;                                 // LDS byte address
    g.z = (unsigned)ga;                            // global_addr[31:0]
    g.w = (unsigned)((ga >> 32) & 0x01FFFFFFu) | (2u << 30);  // addr[56:32],type=2
    return g;
}

__device__ __forceinline__ u32x8 tdm_group1(unsigned tile_rows) {
    const unsigned TD = 1u << 20;   // huge tensor dims: tiles always in-bounds
    u32x8 g;
    g.s0 = 1u << 16;                               // data_size=1 (2 bytes)
    g.s1 = (TD & 0xFFFFu) << 16;                   // tensor_dim0[15:0]
    g.s2 = ((TD >> 16) & 0xFFFFu) | ((TD & 0xFFFFu) << 16);   // d0 hi | d1 lo
    g.s3 = ((TD >> 16) & 0xFFFFu) | (32u << 16);   // d1 hi | tile_dim0=32
    g.s4 = tile_rows;                              // tile_dim1 | tile_dim2=0
    g.s5 = (unsigned)DMODEL;                       // tensor_dim0_stride[31:0]
    g.s6 = 0u;                                     // stride0 hi | stride1 lo
    g.s7 = 0u;                                     // stride1 hi
    return g;
}

__device__ __forceinline__ void tdm_load_to_lds(u32x4 g0, u32x8 g1) {
    asm volatile("tensor_load_to_lds %0, %1" :: "s"(g0), "s"(g1) : "memory");
}

// ---------------------------------------------------------------------------
// Elementwise f32 -> f16 cast
// ---------------------------------------------------------------------------
__global__ void cast_f32_f16_kernel(const float* __restrict__ src,
                                    _Float16* __restrict__ dst, int n) {
    int i = blockIdx.x * blockDim.x + threadIdx.x;
    if (i < n) dst[i] = (_Float16)src[i];
}

// ---------------------------------------------------------------------------
// Transpose + cast: src f32 [2048][2048] -> dst f16 [2048][2048] (dst[c][r])
// ---------------------------------------------------------------------------
__global__ void transpose_cast_kernel(const float* __restrict__ src,
                                      _Float16* __restrict__ dst) {
    __shared__ _Float16 tile[32][33];
    int bx = blockIdx.x * 32;
    int by = blockIdx.y * 32;
    int tx = threadIdx.x, ty = threadIdx.y;   // block 32x8
#pragma unroll
    for (int i = 0; i < 32; i += 8)
        tile[ty + i][tx] = (_Float16)src[(size_t)(by + ty + i) * DMODEL + bx + tx];
    __syncthreads();
#pragma unroll
    for (int i = 0; i < 32; i += 8)
        dst[(size_t)(bx + ty + i) * DMODEL + by + tx] = tile[tx][ty + i];
}

// ---------------------------------------------------------------------------
// WMMA GEMM with TDM double-buffered LDS staging.
// C[4096 x 2048] = A[4096 x 2048] * Bt^T  (Bt stored [N][K] f16)
// Block 256 thr (8 waves), tile 128x256, wave tile 64x64 (4x4 WMMAs/k-step).
// Per k-step wave 0 issues two tensor_load_to_lds (A:128x32, B:256x32),
// tracked with TENSORcnt; compute reads fragments from LDS (ds_load_b128).
// MODE 0: out f32 [M][N] + bias[col]
// MODE 1: out f16 [M][N] * colscale[col]*smul
// MODE 2: out f16 Vt layout [B][H][HDIM][SEQ]
// ---------------------------------------------------------------------------
template <int MODE>
__global__ __launch_bounds__(256)
void gemm_wmma_kernel(const _Float16* __restrict__ A,
                      const _Float16* __restrict__ Bt,
                      void* __restrict__ outp,
                      const float* __restrict__ colscale, float smul,
                      const float* __restrict__ bias) {
    constexpr int K = DMODEL, N = DMODEL;
    constexpr int NSTEP = K / 32;

    __shared__ __align__(16) _Float16 smA[2][128 * 32];   // 2 x 8 KB
    __shared__ __align__(16) _Float16 smB[2][256 * 32];   // 2 x 16 KB

    int lane = threadIdx.x & 31;
    int wv   = threadIdx.x >> 5;
    int rowblk = blockIdx.y * 128;
    int colblk = blockIdx.x * 256;
    int wmRow = (wv >> 2) * 64;       // wave row offset within block tile
    int wnCol = (wv & 3) * 64;        // wave col offset within block tile

    // TDM descriptors (wave-uniform; only wave 0 issues)
    const u32x8 g1A = tdm_group1(128);
    const u32x8 g1B = tdm_group1(256);
    unsigned ldsA[2] = { (unsigned)(uintptr_t)&smA[0][0],
                         (unsigned)(uintptr_t)&smA[1][0] };
    unsigned ldsB[2] = { (unsigned)(uintptr_t)&smB[0][0],
                         (unsigned)(uintptr_t)&smB[1][0] };
    const _Float16* gA = A  + (size_t)rowblk * K;   // + k0 per step
    const _Float16* gB = Bt + (size_t)colblk * K;

    if (wv == 0) {   // prologue: stage k-step 0 into buffer 0
        tdm_load_to_lds(tdm_group0(ldsA[0], gA), g1A);
        tdm_load_to_lds(tdm_group0(ldsB[0], gB), g1B);
    }

    v8f acc[4][4];
#pragma unroll
    for (int i = 0; i < 4; ++i)
#pragma unroll
        for (int j = 0; j < 4; ++j) acc[i][j] = (v8f){};

    for (int t = 0; t < NSTEP; ++t) {
        int cur = t & 1, nxt = cur ^ 1;
        if (wv == 0) {
            if (t + 1 < NSTEP) {   // issue next tile, then wait for current
                int k0 = (t + 1) * 32;
                tdm_load_to_lds(tdm_group0(ldsA[nxt], gA + k0), g1A);
                tdm_load_to_lds(tdm_group0(ldsB[nxt], gB + k0), g1B);
                __builtin_amdgcn_s_wait_tensorcnt(2);
            } else {
                __builtin_amdgcn_s_wait_tensorcnt(0);
            }
        }
        __syncthreads();           // current tiles visible to all waves

        const _Float16* Abuf = &smA[cur][0];
        const _Float16* Bbuf = &smB[cur][0];
        v16h af[4], bf[4];
#pragma unroll
        for (int x = 0; x < 4; ++x) af[x] = load_frag_lds(Abuf, wmRow + x * 16, lane);
#pragma unroll
        for (int x = 0; x < 4; ++x) bf[x] = load_frag_lds(Bbuf, wnCol + x * 16, lane);
#pragma unroll
        for (int mt = 0; mt < 4; ++mt)
#pragma unroll
            for (int nt = 0; nt < 4; ++nt)
                acc[mt][nt] = wmma_f16(af[mt], bf[nt], acc[mt][nt]);

        __syncthreads();           // all reads done before buffer is reused
    }

    int half8 = (lane >> 4) << 3;
#pragma unroll
    for (int nt = 0; nt < 4; ++nt) {
        int col = colblk + wnCol + nt * 16 + (lane & 15);
        float cs = 1.0f;
        if (MODE == 1 && colscale) cs = colscale[col] * smul;
        float bv = (MODE == 0) ? bias[col] : 0.0f;
#pragma unroll
        for (int mt = 0; mt < 4; ++mt) {
#pragma unroll
            for (int r = 0; r < 8; ++r) {
                int row = rowblk + wmRow + mt * 16 + r + half8;
                float v = acc[mt][nt][r];
                if (MODE == 0) {
                    ((float*)outp)[(size_t)row * N + col] = v + bv;
                } else if (MODE == 1) {
                    ((_Float16*)outp)[(size_t)row * N + col] = (_Float16)(v * cs);
                } else {  // MODE 2: V transposed per head [B][H][HDIM][SEQ]
                    int b = row >> 11, s = row & (SEQ - 1);
                    int h = col >> 7,  d = col & (HDIM - 1);
                    ((_Float16*)outp)[((size_t)((b * HEADS + h) * HDIM + d)) * SEQ + s]
                        = (_Float16)v;
                }
            }
        }
    }
}

// ---------------------------------------------------------------------------
// RoPE + head repack: src f16 [B][S][D] (already per-channel scaled)
//   -> dst f16 [B][H][S][HDIM], interleaved-pair rotation.
// ---------------------------------------------------------------------------
__global__ void rope_pack_kernel(const _Float16* __restrict__ src,
                                 _Float16* __restrict__ dst) {
    int tid = blockIdx.x * blockDim.x + threadIdx.x;
    int c2 = tid & (DMODEL / 2 - 1);
    int s  = (tid >> 10) & (SEQ - 1);
    int b  = tid >> 21;
    int ch = c2 * 2;
    int h  = ch >> 7;
    int dh = ch & (HDIM - 1);
    float freq = __expf(-logf(10000.0f) * (float)dh / (float)HDIM);
    float ang  = (float)s * freq;
    float c = cosf(ang), sn = sinf(ang);
    size_t sidx = ((size_t)(b * SEQ + s)) * DMODEL + ch;
    float x0 = (float)src[sidx];
    float x1 = (float)src[sidx + 1];
    size_t didx = (((size_t)(b * HEADS + h) * SEQ + s)) * HDIM + dh;
    dst[didx]     = (_Float16)(x0 * c - x1 * sn);
    dst[didx + 1] = (_Float16)(x1 * c + x0 * sn);
}

// ---------------------------------------------------------------------------
// Flash attention (causal). Grid (SEQ/128, HEADS, BATCH), 256 thr = 8 waves.
// scores^T = K_tile x Q^T (query per lane), online softmax in registers,
// O^T += V^T x P^T with the f32->f16 P conversion needing no lane exchange.
// All 8 K fragments of a key tile are issued as one clause before the score
// WMMAs (staggered loadcnt waits); the 8 V fragments are issued before the
// softmax VALU work so their latency hides behind exp/max/shuffles.
// ---------------------------------------------------------------------------
__global__ __launch_bounds__(256)
void attention_kernel(const _Float16* __restrict__ Qh,
                      const _Float16* __restrict__ Kh,
                      const _Float16* __restrict__ Vt,
                      _Float16* __restrict__ ctx) {
    int lane = threadIdx.x & 31;
    int wv   = threadIdx.x >> 5;
    int h = blockIdx.y, b = blockIdx.z;
    int q0 = blockIdx.x * 128 + wv * 16;

    const _Float16* Qp = Qh + ((size_t)(b * HEADS + h) * SEQ) * HDIM;
    const _Float16* Kp = Kh + ((size_t)(b * HEADS + h) * SEQ) * HDIM;
    const _Float16* Vp = Vt + ((size_t)(b * HEADS + h) * HDIM) * SEQ;

    v16h qb[4];
#pragma unroll
    for (int kc = 0; kc < 4; ++kc) qb[kc] = load_frag(Qp, HDIM, q0, kc * 32, lane);

    v8f o[8];
#pragma unroll
    for (int dc = 0; dc < 8; ++dc) o[dc] = (v8f){};
    float mrun = -1e30f, lrun = 0.0f;
    const float rs = 0.08838834764831845f;    // 1/sqrt(128)
    int qg = q0 + (lane & 15);
    int half8 = (lane >> 4) << 3;

    for (int kb = 0; kb < q0 + 16; kb += 32) {
        // ---- batch all K fragments for this 32-key tile (one load clause) ----
        v16h ka0[4], ka1[4];
#pragma unroll
        for (int kc = 0; kc < 4; ++kc) {
            ka0[kc] = load_frag(Kp, HDIM, kb,      kc * 32, lane);
            ka1[kc] = load_frag(Kp, HDIM, kb + 16, kc * 32, lane);
        }
        v8f s0 = (v8f){}, s1 = (v8f){};
#pragma unroll
        for (int kc = 0; kc < 4; ++kc) {
            s0 = wmma_f16(ka0[kc], qb[kc], s0);
            s1 = wmma_f16(ka1[kc], qb[kc], s1);
        }

        // ---- issue V fragment loads early; latency hides behind softmax ----
        v16h va[8];
#pragma unroll
        for (int dc = 0; dc < 8; ++dc) va[dc] = load_frag(Vp, SEQ, dc * 16, kb, lane);

        // ---- online softmax (per-lane query; keys in regs + lane halves) ----
        float mt = -1e30f;
#pragma unroll
        for (int r = 0; r < 8; ++r) {
            int kg0 = kb + r + half8;
            float v0 = (kg0      <= qg) ? s0[r] * rs : -1e30f;
            float v1 = (kg0 + 16 <= qg) ? s1[r] * rs : -1e30f;
            s0[r] = v0; s1[r] = v1;
            mt = fmaxf(mt, fmaxf(v0, v1));
        }
        mt = fmaxf(mt, __shfl_xor(mt, 16, 32));
        float mnew = fmaxf(mrun, mt);
        float corr = __expf(mrun - mnew);

        float psum = 0.0f;
        v16h pb;
#pragma unroll
        for (int r = 0; r < 8; ++r) {
            float p0 = __expf(s0[r] - mnew);
            float p1 = __expf(s1[r] - mnew);
            psum += p0 + p1;
            pb[r]     = (_Float16)p0;
            pb[r + 8] = (_Float16)p1;
        }
        psum += __shfl_xor(psum, 16, 32);
        lrun = lrun * corr + psum;
        mrun = mnew;

        // ---- O^T += V^T x P^T ----
#pragma unroll
        for (int dc = 0; dc < 8; ++dc) {
            o[dc] = o[dc] * corr;
            o[dc] = wmma_f16(va[dc], pb, o[dc]);
        }
    }

    float invl = 1.0f / lrun;
    size_t rowoff = ((size_t)(b * SEQ + q0 + (lane & 15))) * DMODEL + h * HDIM;
#pragma unroll
    for (int dc = 0; dc < 8; ++dc)
#pragma unroll
        for (int r = 0; r < 8; ++r)
            ctx[rowoff + dc * 16 + r + half8] = (_Float16)(o[dc][r] * invl);
}

// ---------------------------------------------------------------------------
// Launch
// ---------------------------------------------------------------------------
extern "C" void kernel_launch(void* const* d_in, const int* in_sizes, int n_in,
                              void* d_out, int out_size, void* d_ws, size_t ws_size,
                              hipStream_t stream) {
    (void)in_sizes; (void)n_in; (void)out_size; (void)ws_size;
    const float* x  = (const float*)d_in[0];
    const float* Wq = (const float*)d_in[1];
    const float* Wk = (const float*)d_in[2];
    const float* Wv = (const float*)d_in[3];
    const float* Wo = (const float*)d_in[4];
    const float* bo = (const float*)d_in[5];
    const float* sq = (const float*)d_in[6];
    const float* sk = (const float*)d_in[7];
    float* out = (float*)d_out;

    constexpr size_t M1 = 1ull << 20;
    _Float16* ws   = (_Float16*)d_ws;
    _Float16* XH   = ws;             // x cast f16 [B,S,D]      8M
    _Float16* WQT  = ws + 8  * M1;   // W^T [N][K]              4M each
    _Float16* WKT  = ws + 12 * M1;
    _Float16* WVT  = ws + 16 * M1;
    _Float16* WOT  = ws + 20 * M1;
    _Float16* QLIN = ws + 24 * M1;   // scaled Q [B,S,D]        8M
    _Float16* KLIN = ws + 32 * M1;
    _Float16* QH   = ws + 40 * M1;   // rope-packed [B,H,S,HD]
    _Float16* KH   = ws + 48 * M1;
    _Float16* VT   = ws + 56 * M1;   // [B,H,HD,S]
    _Float16* CTX  = ws + 64 * M1;   // [B,S,D]

    const float smul = 45.254833995939045f;  // sqrt(2048)

    {
        int n = BATCH * SEQ * DMODEL;
        cast_f32_f16_kernel<<<(n + 255) / 256, 256, 0, stream>>>(x, XH, n);
    }
    {
        dim3 g(DMODEL / 32, DMODEL / 32), blk(32, 8);
        transpose_cast_kernel<<<g, blk, 0, stream>>>(Wq, WQT);
        transpose_cast_kernel<<<g, blk, 0, stream>>>(Wk, WKT);
        transpose_cast_kernel<<<g, blk, 0, stream>>>(Wv, WVT);
        transpose_cast_kernel<<<g, blk, 0, stream>>>(Wo, WOT);
    }
    {
        dim3 g(DMODEL / 256, (BATCH * SEQ) / 128);
        gemm_wmma_kernel<1><<<g, 256, 0, stream>>>(XH, WQT, QLIN, sq, smul, nullptr);
        gemm_wmma_kernel<1><<<g, 256, 0, stream>>>(XH, WKT, KLIN, sk, smul, nullptr);
        gemm_wmma_kernel<2><<<g, 256, 0, stream>>>(XH, WVT, VT, nullptr, 1.0f, nullptr);
    }
    {
        int n = BATCH * SEQ * DMODEL / 2;
        rope_pack_kernel<<<n / 256, 256, 0, stream>>>(QLIN, QH);
        rope_pack_kernel<<<n / 256, 256, 0, stream>>>(KLIN, KH);
    }
    {
        dim3 g(SEQ / 128, HEADS, BATCH);
        attention_kernel<<<g, 256, 0, stream>>>(QH, KH, VT, CTX);
    }
    {
        dim3 g(DMODEL / 256, (BATCH * SEQ) / 128);
        gemm_wmma_kernel<0><<<g, 256, 0, stream>>>(CTX, WOT, out, nullptr, 1.0f, bo);
    }
}
